// HyperbolicAttention_51118700757528
// MI455X (gfx1250) — matrix operand order repaired and checked by reading, not compiled, not run on hardware
//
#include <hip/hip_runtime.h>
#include <hip/hip_bf16.h>
#include <math.h>

#define N_NODES 100000
#define N_EDGES 1600000
#define HID 128
#define EPSF 1e-5f

// LDS row stride for the 32x256 att_in tile, padded to kill bank conflicts:
// 264 halves = 132 dwords -> lane l15 of an A b128 load hits banks 4*l15..4*l15+3.
#define RSTR 264

typedef __attribute__((ext_vector_type(16))) _Float16 v16h;
typedef __attribute__((ext_vector_type(8)))  float    v8f;

// ---------- monotonic float <-> uint key (for atomic segment max) ----------
__device__ __forceinline__ unsigned fkey(float f) {
    unsigned b = __float_as_uint(f);
    return b ^ ((unsigned)((int)b >> 31) | 0x80000000u);
}
__device__ __forceinline__ float unfkey(unsigned k) {
    unsigned b = (k & 0x80000000u) ? (k ^ 0x80000000u) : ~k;
    return __uint_as_float(b);
}

// ---------- kernel 0: init segment max/sum ----------
__global__ void k_init(unsigned* __restrict__ maxkey, float* __restrict__ sums, int n) {
    int i = blockIdx.x * blockDim.x + threadIdx.x;
    if (i < n) { maxkey[i] = 0u; sums[i] = 0.0f; }
}

// ---------- kernel 1: log_map(h_i) -> f16, one wave per row ----------
__global__ __launch_bounds__(256) void k_node_logmap(const float* __restrict__ h,
                                                     const float* __restrict__ curv,
                                                     _Float16* __restrict__ out, int n_rows) {
    int lane = threadIdx.x & 31;
    int wave = blockIdx.x * (blockDim.x >> 5) + (threadIdx.x >> 5);
    int nw   = gridDim.x * (blockDim.x >> 5);
    float kk = fminf(curv[0], -1e-5f);
    float sqrtk = sqrtf(-kk);
    for (int r = wave; r < n_rows; r += nw) {
        float4 x = *reinterpret_cast<const float4*>(h + (size_t)r * HID + lane * 4);
        float ss = x.x*x.x + x.y*x.y + x.z*x.z + x.w*x.w;
        #pragma unroll
        for (int m = 16; m >= 1; m >>= 1) ss += __shfl_xor(ss, m, 32);
        float norm = sqrtf(ss);
        float xn = fmaxf(norm, EPSF);
        float z  = fminf(norm, 1.0f - EPSF);
        float scale = atanhf(z) / (sqrtk * xn);
        union { _Float16 hh[4]; uint2 u; } pk;
        pk.hh[0] = (_Float16)(x.x * scale);
        pk.hh[1] = (_Float16)(x.y * scale);
        pk.hh[2] = (_Float16)(x.z * scale);
        pk.hh[3] = (_Float16)(x.w * scale);
        *reinterpret_cast<uint2*>(out + (size_t)r * HID + lane * 4) = pk.u;
    }
}

// ---------- kernel 2: W1 f32 -> f16 ----------
__global__ void k_w1cvt(const float* __restrict__ w1, _Float16* __restrict__ w1h, int n) {
    int i = blockIdx.x * blockDim.x + threadIdx.x;
    if (i < n) w1h[i] = (_Float16)w1[i];
}

// ---------- kernel 3: fused edge attention logits (WMMA GEMM) ----------
// Block = 256 threads = 8 waves. Wave w owns output n-tile w (cols 16w..16w+15).
// Per iteration: 32 edges = two 16-row M-tiles sharing the VGPR-resident W1
// B-fragments; two independent WMMA accumulation chains per wave.
__global__ __launch_bounds__(256) void k_edge_attn(
    const _Float16* __restrict__ hitan, const float* __restrict__ hnb,
    const float* __restrict__ curv, const _Float16* __restrict__ w1h,
    const float* __restrict__ b1, const float* __restrict__ w2,
    const float* __restrict__ b2, const int* __restrict__ row,
    float* __restrict__ logits, unsigned* __restrict__ maxkey, int n_iters) {

    __shared__ _Float16 attLDS[32 * RSTR];  // ~16.9 KB, padded rows
    __shared__ float    partial[8][32];     // per-wave per-row partial dot
    __shared__ int      rowIdx[32];

    const int tid  = threadIdx.x;
    const int w    = tid >> 5;        // wave id 0..7 == n-tile
    const int lane = tid & 31;
    const int l15  = lane & 15;
    const int hi   = lane >> 4;       // 0 for lanes 0-15, 1 for 16-31

    const float kk = fminf(curv[0], -1e-5f);
    const float sqrtk = sqrtf(-kk);
    const float b2v = b2[0];
    const int   n   = w * 16 + l15;   // output column this lane owns
    const float b1v = b1[n];
    const float w2v = w2[n];

    // ---- preload B fragments of W1 into VGPRs (contiguous k order, same as the A loads) ----
    // lane 0-15: N=l15, K = kt*32 + 0..15 ; lane 16-31: N=l15, K = kt*32 + 16..31
    const int khiB = hi * 16;
    v16h bfrag[8];
    #pragma unroll
    for (int kt = 0; kt < 8; ++kt) {
        union { v16h v; _Float16 hh[16]; } B;
        #pragma unroll
        for (int i = 0; i < 16; ++i) {
            int k = kt * 32 + khiB + i;
            B.hh[i] = w1h[k * HID + n];
        }
        bfrag[kt] = B.v;
    }

    const int koffA = hi * 16;  // A layout: lanes>=16 hold K+16..31 block

    for (int it = blockIdx.x; it < n_iters; it += gridDim.x) {
        const int e0 = it * 32;
        __syncthreads();
        if (tid < 32) rowIdx[tid] = row[e0 + tid];
        __syncthreads();

        // ---- fill att_in tile (32 edges): wave w handles edges 4w..4w+3 ----
        #pragma unroll
        for (int mm = 0; mm < 4; ++mm) {
            const int m = 4 * w + mm;
            const int e = e0 + m;
            const int r = rowIdx[m];
            // gather precomputed log_map(h_i) (f16)
            uint2 g = *reinterpret_cast<const uint2*>(hitan + (size_t)r * HID + lane * 4);
            *reinterpret_cast<uint2*>(&attLDS[m * RSTR + lane * 4]) = g;
            // inline log_map(h_neighbors[e])
            float4 x = *reinterpret_cast<const float4*>(hnb + (size_t)e * HID + lane * 4);
            float ss = x.x*x.x + x.y*x.y + x.z*x.z + x.w*x.w;
            #pragma unroll
            for (int msk = 16; msk >= 1; msk >>= 1) ss += __shfl_xor(ss, msk, 32);
            float norm = sqrtf(ss);
            float xn = fmaxf(norm, EPSF);
            float z  = fminf(norm, 1.0f - EPSF);
            float scale = atanhf(z) / (sqrtk * xn);
            union { _Float16 hh[4]; uint2 u; } pk;
            pk.hh[0] = (_Float16)(x.x * scale);
            pk.hh[1] = (_Float16)(x.y * scale);
            pk.hh[2] = (_Float16)(x.z * scale);
            pk.hh[3] = (_Float16)(x.w * scale);
            *reinterpret_cast<uint2*>(&attLDS[m * RSTR + 128 + lane * 4]) = pk.u;
            // prefetch next iteration's h_neighbors (gfx1250 global_prefetch_b8)
            if (it + gridDim.x < n_iters) {
                const float* pf = hnb + (size_t)(e + (size_t)gridDim.x * 32) * HID + lane * 4;
                __builtin_prefetch(pf, 0, 1);
            }
        }
        __syncthreads();

        // ---- GEMM: 8 k-steps x 2 M-tiles, independent accumulator chains ----
        v8f acc0 = {};
        v8f acc1 = {};
        const int aBase0 = l15 * RSTR;          // rows 0..15
        const int aBase1 = (16 + l15) * RSTR;   // rows 16..31
        #pragma unroll
        for (int kt = 0; kt < 8; ++kt) {
            union { v16h v; uint4 u[2]; } A0, A1;
            A0.u[0] = *reinterpret_cast<const uint4*>(&attLDS[aBase0 + kt * 32 + koffA]);
            A0.u[1] = *reinterpret_cast<const uint4*>(&attLDS[aBase0 + kt * 32 + 8 + koffA]);
            A1.u[0] = *reinterpret_cast<const uint4*>(&attLDS[aBase1 + kt * 32 + koffA]);
            A1.u[1] = *reinterpret_cast<const uint4*>(&attLDS[aBase1 + kt * 32 + 8 + koffA]);
            acc0 = __builtin_amdgcn_wmma_f32_16x16x32_f16(
                false, A0.v, false, bfrag[kt], (short)0, acc0, false, false);
            acc1 = __builtin_amdgcn_wmma_f32_16x16x32_f16(
                false, A1.v, false, bfrag[kt], (short)0, acc1, false, false);
        }

        // ---- epilogue: bias + relu + dot(W2), half-wave reduce per M-tile ----
        // acc_t[r]: hdn[m = t*16 + r + 8*hi][n], n = 16w + l15
        #pragma unroll
        for (int t = 0; t < 2; ++t) {
            float p[8];
            #pragma unroll
            for (int r = 0; r < 8; ++r) {
                float a = (t == 0) ? acc0[r] : acc1[r];
                p[r] = fmaxf(a + b1v, 0.0f) * w2v;
            }
            #pragma unroll
            for (int r = 0; r < 8; ++r) {
                p[r] += __shfl_xor(p[r], 1, 32);
                p[r] += __shfl_xor(p[r], 2, 32);
                p[r] += __shfl_xor(p[r], 4, 32);
                p[r] += __shfl_xor(p[r], 8, 32);
            }
            if (l15 < 8) {
                #pragma unroll
                for (int r = 0; r < 8; ++r)
                    if (l15 == r) partial[w][t * 16 + hi * 8 + r] = p[r];
            }
        }
        __syncthreads();

        if (tid < 32) {
            float s = b2v;
            #pragma unroll
            for (int ww = 0; ww < 8; ++ww) s += partial[ww][tid];
            logits[e0 + tid] = s;
            atomicMax(&maxkey[rowIdx[tid]], fkey(s));
        }
    }
}

// ---------- kernel 4: e = exp(logit - max[row]); segment sum ----------
__global__ void k_expsum(float* __restrict__ logits, const int* __restrict__ row,
                         const unsigned* __restrict__ maxkey, float* __restrict__ sums, int nE) {
    int stride = gridDim.x * blockDim.x;
    for (int e = blockIdx.x * blockDim.x + threadIdx.x; e < nE; e += stride) {
        int r = row[e];
        float m = unfkey(maxkey[r]);
        float ev = __expf(logits[e] - m);
        logits[e] = ev;                 // reuse buffer for exp values
        atomicAdd(&sums[r], ev);
    }
}

// ---------- kernel 5: normalize ----------
__global__ void k_norm(const float* __restrict__ evals, const int* __restrict__ row,
                       const float* __restrict__ sums, float* __restrict__ out, int nE) {
    int stride = gridDim.x * blockDim.x;
    for (int e = blockIdx.x * blockDim.x + threadIdx.x; e < nE; e += stride)
        out[e] = evals[e] / sums[row[e]];
}

extern "C" void kernel_launch(void* const* d_in, const int* in_sizes, int n_in,
                              void* d_out, int out_size, void* d_ws, size_t ws_size,
                              hipStream_t stream) {
    const float* h_i  = (const float*)d_in[0];
    const float* h_nb = (const float*)d_in[1];
    const float* curv = (const float*)d_in[2];
    const float* W1   = (const float*)d_in[3];
    const float* b1   = (const float*)d_in[4];
    const float* W2   = (const float*)d_in[5];
    const float* b2   = (const float*)d_in[6];
    const int*   row  = (const int*)d_in[7];   // edge_index[0] = first N_EDGES ints
    float* out = (float*)d_out;

    char* ws = (char*)d_ws;
    size_t off = 0;
    auto carve = [&](size_t bytes) -> void* {
        void* p = ws + off;
        off = (off + bytes + 255) & ~(size_t)255;
        return p;
    };
    _Float16* hitan  = (_Float16*)carve((size_t)N_NODES * HID * 2);
    _Float16* w1h    = (_Float16*)carve((size_t)2 * HID * HID * 2);
    float*    logits = (float*)   carve((size_t)N_EDGES * 4);
    unsigned* maxkey = (unsigned*)carve((size_t)N_NODES * 4);
    float*    sums   = (float*)   carve((size_t)N_NODES * 4);

    // 0: init segment max/sum (deterministic per call)
    k_init<<<(N_NODES + 255) / 256, 256, 0, stream>>>(maxkey, sums, N_NODES);
    // 1: node log_map -> f16
    k_node_logmap<<<1024, 256, 0, stream>>>(h_i, curv, hitan, N_NODES);
    // 2: W1 -> f16
    k_w1cvt<<<(2 * HID * HID + 255) / 256, 256, 0, stream>>>(W1, w1h, 2 * HID * HID);
    // 3: fused WMMA edge-attention logits + segment max (32 edges / iter)
    const int n_iters = N_EDGES / 32;
    k_edge_attn<<<4096, 256, 0, stream>>>(hitan, h_nb, curv, w1h, b1, W2, b2,
                                          row, logits, maxkey, n_iters);
    // 4: exp + segment sum
    k_expsum<<<2048, 256, 0, stream>>>(logits, row, maxkey, sums, N_EDGES);
    // 5: normalize
    k_norm<<<2048, 256, 0, stream>>>(logits, row, sums, out, N_EDGES);
}
